// RNN_scan_23854248362163
// MI455X (gfx1250) — matrix operand order repaired and checked by reading, compiled, not run on hardware
//
#include <hip/hip_runtime.h>
#include <hip/hip_bf16.h>

// Shapes from the reference
#define T_STEPS 1024
#define BATCH   64
#define DIM     1024          // D_IN == H == 1024
#define MROWS   (T_STEPS * BATCH)   // 65536 rows of xs
#define NWG_SCAN 32

typedef __attribute__((ext_vector_type(16))) __bf16 v16bf;
typedef __attribute__((ext_vector_type(8)))  __bf16 v8bf;
typedef __attribute__((ext_vector_type(8)))  float  v8f;
typedef __attribute__((ext_vector_type(4)))  unsigned int v4u;
typedef __attribute__((ext_vector_type(8)))  int v8i;
typedef __attribute__((ext_vector_type(4)))  int v4i;

static __device__ __forceinline__ v16bf join_bf(v8bf lo, v8bf hi) {
  v16bf r;
#pragma unroll
  for (int i = 0; i < 8; ++i) { r[i] = lo[i]; r[i + 8] = hi[i]; }
  return r;
}

static __device__ __forceinline__ v8f zero_v8f() {
  v8f z;
#pragma unroll
  for (int i = 0; i < 8; ++i) z[i] = 0.0f;
  return z;
}

// ---------------------------------------------------------------------------
// prep: Whh (f32, [K][N]) -> WhhT (bf16, [N][K]); init -> h bf16 buffer 0;
//       reset grid-barrier words (ws is poisoned; re-init every call).
// ---------------------------------------------------------------------------
__global__ void prep_kernel(const float* __restrict__ Whh,
                            const float* __restrict__ initH,
                            __bf16* __restrict__ WhhT,
                            __bf16* __restrict__ hbuf0,
                            unsigned* __restrict__ bar) {
  size_t tid = (size_t)blockIdx.x * blockDim.x + threadIdx.x;
  if (tid < 2) bar[tid] = 0u;
  if (tid < (size_t)DIM * DIM) {
    int k = (int)(tid >> 10);
    int n = (int)(tid & (DIM - 1));
    WhhT[(size_t)n * DIM + k] = (__bf16)Whh[tid];
  }
  if (tid < (size_t)BATCH * DIM) {
    hbuf0[tid] = (__bf16)initH[tid];
  }
}

// ---------------------------------------------------------------------------
// Phase 1: xproj[t,b,:] = xs[t,b,:] @ Wih + bih  ->  outs region of d_out.
// Tiled bf16 WMMA GEMM: BM=128, BN=128, BK=32, 256 threads (8 waves),
// wave grid 4(M) x 2(N), each wave owns 2x4 16x16 accumulators.
// ---------------------------------------------------------------------------
__global__ void __launch_bounds__(256)
xproj_gemm_kernel(const float* __restrict__ xs,
                  const float* __restrict__ Wih,
                  const float* __restrict__ bih,
                  float* __restrict__ outs) {
  __shared__ __bf16 As[128 * 32];   // [row][k], row-major, 8 KB
  __shared__ __bf16 Bst[128 * 32];  // [n][k]   (B transposed), 8 KB

  const int tidx = threadIdx.x;
  const int lane = tidx & 31;
  const int wave = tidx >> 5;
  const int wm = wave & 3;     // 0..3 -> 32 rows each
  const int wn = wave >> 2;    // 0..1 -> 64 cols each
  const int lr = lane & 15;    // row/col within a 16x16 tile
  const int lh = lane >> 4;    // lane half (WMMA operand layout)

  const int blockN = blockIdx.x * 128;
  const int blockM = blockIdx.y * 128;

  v8f c[2][4];
#pragma unroll
  for (int im = 0; im < 2; ++im)
#pragma unroll
    for (int in = 0; in < 4; ++in) c[im][in] = zero_v8f();

  for (int kt = 0; kt < DIM / 32; ++kt) {
    const int k0 = kt * 32;
    // --- stage A tile (128x32 f32 -> bf16), float4 per thread, 4 iters ---
#pragma unroll
    for (int i = 0; i < 4; ++i) {
      int slot = tidx + i * 256;          // 0..1023
      int r = slot >> 3;                  // 0..127
      int c4 = (slot & 7) * 4;            // 0..28
      const float4 v = *(const float4*)(xs + (size_t)(blockM + r) * DIM + k0 + c4);
      __bf16* dst = &As[r * 32 + c4];
      dst[0] = (__bf16)v.x; dst[1] = (__bf16)v.y;
      dst[2] = (__bf16)v.z; dst[3] = (__bf16)v.w;
    }
    // --- stage B tile (32x128 f32 -> bf16, transposed into [n][k]) ---
#pragma unroll
    for (int i = 0; i < 4; ++i) {
      int slot = tidx + i * 256;          // 0..1023
      int k = slot >> 5;                  // 0..31
      int n4 = (slot & 31) * 4;           // 0..124
      const float4 v = *(const float4*)(Wih + (size_t)(k0 + k) * DIM + blockN + n4);
      Bst[(n4 + 0) * 32 + k] = (__bf16)v.x;
      Bst[(n4 + 1) * 32 + k] = (__bf16)v.y;
      Bst[(n4 + 2) * 32 + k] = (__bf16)v.z;
      Bst[(n4 + 3) * 32 + k] = (__bf16)v.w;
    }
    __syncthreads();

    // --- fragments + 8 WMMAs ---
    v16bf a[2], b[4];
#pragma unroll
    for (int im = 0; im < 2; ++im) {
      const int row = wm * 32 + im * 16 + lr;
      const v8bf lo = *(const v8bf*)&As[row * 32 + lh * 8];
      const v8bf hi = *(const v8bf*)&As[row * 32 + 16 + lh * 8];
      a[im] = join_bf(lo, hi);
    }
#pragma unroll
    for (int in = 0; in < 4; ++in) {
      const int col = wn * 64 + in * 16 + lr;
      const v8bf lo = *(const v8bf*)&Bst[col * 32 + lh * 8];
      const v8bf hi = *(const v8bf*)&Bst[col * 32 + 16 + lh * 8];
      b[in] = join_bf(lo, hi);
    }
#pragma unroll
    for (int im = 0; im < 2; ++im)
#pragma unroll
      for (int in = 0; in < 4; ++in)
        c[im][in] = __builtin_amdgcn_wmma_f32_16x16x32_bf16(
            false, a[im], false, b[in], (short)0, c[im][in], false, false);
    __syncthreads();
  }

  // --- epilogue: + bih, store f32 ---
#pragma unroll
  for (int im = 0; im < 2; ++im) {
#pragma unroll
    for (int in = 0; in < 4; ++in) {
      const int col = blockN + wn * 64 + in * 16 + lr;
      const float bi = bih[col];
      const int rbase = blockM + wm * 32 + im * 16 + lh * 8;
#pragma unroll
      for (int e = 0; e < 8; ++e)
        outs[(size_t)(rbase + e) * DIM + col] = c[im][in][e] + bi;
    }
  }
}

// ---------------------------------------------------------------------------
// Grid-wide barrier across NWG_SCAN resident workgroups (two-word, gen-count)
// ---------------------------------------------------------------------------
static __device__ __forceinline__ void grid_barrier(unsigned* cnt, unsigned* gen,
                                                    unsigned nwg) {
  __threadfence();          // make this thread's writes device-visible
  __syncthreads();          // everyone in the WG done (stores issued + fenced)
  if (threadIdx.x == 0) {
    unsigned g = __hip_atomic_load(gen, __ATOMIC_ACQUIRE, __HIP_MEMORY_SCOPE_AGENT);
    unsigned old = __hip_atomic_fetch_add(cnt, 1u, __ATOMIC_ACQ_REL,
                                          __HIP_MEMORY_SCOPE_AGENT);
    if (old == nwg - 1u) {
      __hip_atomic_store(cnt, 0u, __ATOMIC_RELAXED, __HIP_MEMORY_SCOPE_AGENT);
      __hip_atomic_fetch_add(gen, 1u, __ATOMIC_RELEASE, __HIP_MEMORY_SCOPE_AGENT);
    } else {
      while (__hip_atomic_load(gen, __ATOMIC_ACQUIRE, __HIP_MEMORY_SCOPE_AGENT) == g)
        __builtin_amdgcn_s_sleep(1);
    }
  }
  __syncthreads();
  __threadfence();
}

// ---------------------------------------------------------------------------
// Phase 2: persistent scan. 32 WGs x 8 waves = 256 waves = 4(M) x 64(N) tiles.
// Per step, per wave: 32 accumulating v_wmma_f32_16x16x32_bf16 over K=1024.
// B panel (32 columns of Whh, bf16, 64 KB) staged once in LDS via the
// Tensor Data Mover (tensor_load_to_lds + s_wait_tensorcnt); h ping-pongs
// in bf16; outs updated in place (xproj[t] consumed -> h[t] written).
// ---------------------------------------------------------------------------
__global__ void __launch_bounds__(256)
scan_kernel(float* __restrict__ outs,        // [T][B][H], holds xproj on entry
            float* __restrict__ carry,       // [B][H]
            const float* __restrict__ bhh,   // [H]
            __bf16* __restrict__ hbuf,       // [2][B*H] bf16 ping-pong
            const __bf16* __restrict__ WhhT, // [N=H][K=H] bf16
            unsigned* __restrict__ bar) {
  __shared__ __bf16 Bs[32 * DIM];  // 64 KB: 32 columns of Whh, all K.
                                   // Only LDS object in this kernel -> LDS offset 0.

  const int lane = threadIdx.x & 31;
  const int wave = threadIdx.x >> 5;
  const int gw = blockIdx.x * 8 + wave;  // 0..255
  const int mtile = gw & 3;              // 0..3   (M = 64)
  const int ntile = gw >> 2;             // 0..63  (N = 1024)
  const int m0 = mtile * 16;
  const int n0 = ntile * 16;
  const int lr = lane & 15;
  const int lh = lane >> 4;

  // --- TDM: stage this WG's 32 rows of WhhT (= 32 Whh columns) into LDS ---
  const int colBase = blockIdx.x * 32;
  if (threadIdx.x < 32) {  // wave 0 only (uniform branch; TDM ignores EXEC)
    const unsigned long long ga =
        (unsigned long long)(const void*)(WhhT + (size_t)colBase * DIM);
    // D# group 0: count=1 | lds_addr | global_addr | type=2 ("image")
    v4u g0;
    g0[0] = 1u;                                   // count=1, user descriptor
    g0[1] = 0u;                                   // lds_addr (Bs at offset 0)
    g0[2] = (unsigned)(ga & 0xFFFFFFFFu);         // global_addr[31:0]
    g0[3] = (unsigned)((ga >> 32) & 0x01FFFFFFu)  // global_addr[56:32]
            | 0x80000000u;                        // type=2 -> bits[127:126]=10b
    // D# group 1: 2D tile, data_size=2B, tile 1024(x) x 32(y), stride 1024
    v8i g1;
    g1[0] = 0x00010000;                 // data_size=1 (2 bytes); no mask/flags
    g1[1] = (int)((DIM & 0xFFFF) << 16);          // tensor_dim0[15:0]=1024
    g1[2] = (int)((DIM & 0xFFFF) << 16);          // dim0 hi=0 | tensor_dim1 lo=1024
    g1[3] = (int)((DIM & 0xFFFF) << 16);          // dim1 hi=0 | tile_dim0=1024
    g1[4] = 32;                                   // tile_dim1=32, tile_dim2=0
    g1[5] = DIM;                                  // tensor_dim0_stride lo32
    g1[6] = 0;                                    // stride hi | dim1_stride lo
    g1[7] = 0;
    v4i gz4 = {0, 0, 0, 0};                       // groups 2/3 unused (2D tile)
    v8i gz8 = {0, 0, 0, 0, 0, 0, 0, 0};           // trailing group (unused)
    __builtin_amdgcn_tensor_load_to_lds(g0, g1, gz4, gz4, gz8, 0);
    __builtin_amdgcn_s_wait_tensorcnt(0);
  }
  __syncthreads();

  const int col = n0 + lr;            // output column of this lane
  const int ldsCol = col - colBase;   // 0..31
  const float bh = bhh[col];
  const int crow = m0 + lh * 8;       // first C row of this lane
  const int arow = m0 + lr;           // A row of this lane
  const int abase = lh * 8;           // A k sub-offset
  const int bbase = lh * 8;           // B k sub-offset

  for (int t = 0; t < T_STEPS; ++t) {
    const __bf16* __restrict__ hb = hbuf + (size_t)(t & 1) * (BATCH * DIM);
    __bf16* __restrict__ hn = hbuf + (size_t)((t + 1) & 1) * (BATCH * DIM);
    float* __restrict__ xp = outs + (size_t)t * (BATCH * DIM);

    // prefetch next step's xproj lines while this step's WMMA chain runs
    if (t + 1 < T_STEPS)
      __builtin_prefetch(xp + (size_t)BATCH * DIM + (size_t)crow * DIM + col, 0, 1);

    // C init = xproj + bhh
    v8f c;
#pragma unroll
    for (int e = 0; e < 8; ++e) c[e] = xp[(size_t)(crow + e) * DIM + col] + bh;

    // K accumulation: 32 chained bf16 WMMAs
    const __bf16* aRow = hb + (size_t)arow * DIM;
    const __bf16* bRow = Bs + (size_t)ldsCol * DIM;
#pragma unroll 4
    for (int kc = 0; kc < DIM / 32; ++kc) {
      const int kb = kc * 32;
      const v8bf alo = *(const v8bf*)(aRow + kb + abase);
      const v8bf ahi = *(const v8bf*)(aRow + kb + 16 + abase);
      const v8bf blo = *(const v8bf*)(bRow + kb + bbase);
      const v8bf bhi = *(const v8bf*)(bRow + kb + 16 + bbase);
      c = __builtin_amdgcn_wmma_f32_16x16x32_bf16(
          false, join_bf(alo, ahi), false, join_bf(blo, bhi),
          (short)0, c, false, false);
    }

    // activation + stores (f32 outs in place, bf16 next-h)
#pragma unroll
    for (int e = 0; e < 8; ++e) {
      const float hv = tanhf(c[e]);
      const size_t idx = (size_t)(crow + e) * DIM + col;
      xp[idx] = hv;
      hn[idx] = (__bf16)hv;
      if (t == T_STEPS - 1) carry[idx] = hv;
    }

    grid_barrier(bar, bar + 1, NWG_SCAN);
  }
}

// ---------------------------------------------------------------------------
extern "C" void kernel_launch(void* const* d_in, const int* in_sizes, int n_in,
                              void* d_out, int out_size, void* d_ws, size_t ws_size,
                              hipStream_t stream) {
  const float* initH = (const float*)d_in[0];  // [B,H]
  const float* xs    = (const float*)d_in[1];  // [T,B,D]
  const float* Wih   = (const float*)d_in[2];  // [D,H]
  const float* bih   = (const float*)d_in[3];  // [H]
  const float* Whh   = (const float*)d_in[4];  // [H,H]
  const float* bhh   = (const float*)d_in[5];  // [H]

  float* carry = (float*)d_out;                 // [B,H]
  float* outs  = carry + (size_t)BATCH * DIM;   // [T,B,H]

  // workspace layout (~2.3 MB): barrier | hbuf ping-pong | WhhT
  char* ws = (char*)d_ws;
  unsigned* bar = (unsigned*)ws;                                   // 2 u32
  __bf16* hbuf  = (__bf16*)(ws + 256);                             // 2*64K bf16
  __bf16* WhhT  = (__bf16*)(ws + 256 + 2u * BATCH * DIM * sizeof(__bf16));

  // 1) one-time converts + barrier reset
  prep_kernel<<<(DIM * DIM + 255) / 256, 256, 0, stream>>>(Whh, initH, WhhT, hbuf, bar);

  // 2) xproj = xs @ Wih + bih  (bf16 WMMA, f32 accumulate) -> outs
  dim3 g1(DIM / 128, MROWS / 128);  // (8, 512)
  xproj_gemm_kernel<<<g1, 256, 0, stream>>>(xs, Wih, bih, outs);

  // 3) persistent recurrent scan
  scan_kernel<<<NWG_SCAN, 256, 0, stream>>>(outs, carry, bhh, hbuf, WhhT, bar);
}
